// SimpleRNN_70849780514822
// MI455X (gfx1250) — compile-verified, matrix operand order checked by reading, not run on hardware
//
#include <hip/hip_runtime.h>

// ---------- types ----------
typedef __attribute__((ext_vector_type(16))) __bf16 v16bf;
typedef __attribute__((ext_vector_type(8)))  float  v8f;
typedef __attribute__((ext_vector_type(4)))  unsigned v4u;

#define TT   512
#define BB   64
#define DD   512   // D_IN == D_HID == D_OUT == 512

// Load one wave-lane's A/B fragment for v_wmma_f32_16x16x32_bf16.
// Layout (16-bit A 16x32, ISA 7.12.2): lanes 0-15 hold K[0..7] (v0..3) and
// K[16..23] (v4..7); lanes 16-31 hold K[8..15] and K[24..31]. With the caller
// pre-offsetting by (lane>>4)*8 elements, that is two contiguous 16B loads.
__device__ __forceinline__ v16bf load_frag(const __bf16* p) {
  union { v4u u[2]; v16bf v; } t;
  t.u[0] = *(const v4u*)(p);
  t.u[1] = *(const v4u*)(p + 16);
  return t.v;
}

__device__ __forceinline__ v8f wmma_bf16(v16bf a, v16bf b, v8f c) {
  return __builtin_amdgcn_wmma_f32_16x16x32_bf16(
      /*neg_a=*/false, a, /*neg_b=*/false, b,
      /*c_mod=*/(short)0, c, /*reuse_a=*/false, /*reuse_b=*/false);
}

// ---------- prep kernels ----------

// dst[n*512+k] = bf16(src[k*512+n])  (transpose 512x512 f32 -> bf16)
__global__ void cvt_transpose512(const float* __restrict__ src, __bf16* __restrict__ dst) {
  int idx = blockIdx.x * 256 + threadIdx.x;     // n*512 + k
  int n = idx >> 9, k = idx & 511;
  dst[idx] = (__bf16)src[k * DD + n];
}

__global__ void cvt_bf16(const float* __restrict__ src, __bf16* __restrict__ dst, int n) {
  int i = blockIdx.x * 256 + threadIdx.x;
  if (i < n) dst[i] = (__bf16)src[i];
}

// broadcast h0 (1x512) to (64x512) bf16
__global__ void bcast_h0(const float* __restrict__ h0, __bf16* __restrict__ dst) {
  int i = blockIdx.x * 256 + threadIdx.x;       // 0..32767
  dst[i] = (__bf16)h0[i & (DD - 1)];
}

// ---------- C/D tile store (f32 16x16 layout: lanes 0-15 vgpr rr -> (M=rr,N=lane);
//            lanes 16-31 -> (M=8+rr, N=lane-16)) ----------
template <bool OUT_BF16>
__device__ __forceinline__ void store_tile(v8f acc, int mt, int nt,
                                           const float* __restrict__ bias,
                                           void* __restrict__ outv, int r, int hk) {
  const int col  = nt * 16 + r;
  const int mrow = mt * 16 + hk * 8;
  const float bv = bias ? bias[col] : 0.0f;
#pragma unroll
  for (int rr = 0; rr < 8; ++rr) {
    float v = acc[rr] + bv;
    if (OUT_BF16)
      ((__bf16*)outv)[(size_t)(mrow + rr) * DD + col] = (__bf16)v;
    else
      ((float*)outv)[(size_t)(mrow + rr) * DD + col] = v;
  }
}

// ---------- big parallel GEMM: C[M,512] = A[M,512] @ Bt[512,512]^T (+bias) ----------
// Bt is stored [N][K] (pre-transposed) so B fragments load like A fragments.
// Register blocking: one wave computes a 32x32 output block (4 WMMA tiles),
// reusing each A/B fragment twice -> 16 FLOP/byte from L2.
template <bool OUT_BF16>
__global__ void __launch_bounds__(256)
gemm512_bf16(const __bf16* __restrict__ A, const __bf16* __restrict__ Bt,
             const float* __restrict__ bias, void* __restrict__ outv) {
  const int lane = threadIdx.x & 31;
  const int w    = blockIdx.x * 8 + (threadIdx.x >> 5);
  const int mt   = (w >> 4) << 1;   // base 16-row tile (covers mt, mt+1)
  const int nt   = (w & 15) << 1;   // base 16-col tile (covers nt, nt+1)
  const int r    = lane & 15;
  const int hk   = lane >> 4;

  const __bf16* ap0 = A  + (size_t)(mt * 16 + r) * DD + hk * 8;
  const __bf16* ap1 = ap0 + 16 * DD;
  const __bf16* bp0 = Bt + (size_t)(nt * 16 + r) * DD + hk * 8;
  const __bf16* bp1 = bp0 + 16 * DD;

  v8f acc00 = {}, acc01 = {}, acc10 = {}, acc11 = {};
#pragma unroll 2
  for (int k = 0; k < DD; k += 32) {
    v16bf a0 = load_frag(ap0 + k);
    v16bf a1 = load_frag(ap1 + k);
    v16bf b0 = load_frag(bp0 + k);
    v16bf b1 = load_frag(bp1 + k);
    // keep the 8 loads grouped above the 4 WMMAs (pipelined waits)
    __builtin_amdgcn_sched_barrier(0);
    acc00 = wmma_bf16(a0, b0, acc00);
    acc01 = wmma_bf16(a0, b1, acc01);
    acc10 = wmma_bf16(a1, b0, acc10);
    acc11 = wmma_bf16(a1, b1, acc11);
  }

  store_tile<OUT_BF16>(acc00, mt,     nt,     bias, outv, r, hk);
  store_tile<OUT_BF16>(acc01, mt,     nt + 1, bias, outv, r, hk);
  store_tile<OUT_BF16>(acc10, mt + 1, nt,     bias, outv, r, hk);
  store_tile<OUT_BF16>(acc11, mt + 1, nt + 1, bias, outv, r, hk);
}

// ---------- persistent recurrence kernel ----------
// 128 blocks x 64 threads; one block per 16x16 tile of the 64x512 h matrix.
// Split-K: wave w of each block handles K in [w*256, w*256+256):
//   - bfrag[8] (64 VGPRs) of hh register-resident for all 512 steps
//   - afrag[8] (64 VGPRs) batched each step; sched_barrier(0) pins the 16
//     b128 loads ABOVE the WMMA chain -> one pipelined L2 round trip
//   - wave 1 -> partial tile via LDS -> wave 0 combines + XW + relu + store
//   - XW for step t+1 prefetched before the barrier (XW is precomputed/const)
// Wave (mt,nt) only touches h rows [mt*16, mt*16+16): sync per mt-group
// (32 blocks) via 4 independent padded counters.
__global__ void __launch_bounds__(64)
rnn_steps(const __bf16* __restrict__ hhT, const __bf16* __restrict__ XWb,
          const __bf16* __restrict__ h0b, __bf16* __restrict__ H,
          unsigned* __restrict__ counters) {
  __shared__ v8f part[32];

  const int lane  = threadIdx.x & 31;
  const int widx  = threadIdx.x >> 5;     // K-half owner (wave-uniform)
  const int mt    = blockIdx.x >> 5;      // 0..3
  const int nt    = blockIdx.x & 31;      // 0..31
  const int r     = lane & 15;
  const int hk    = lane >> 4;
  const int kbase = widx * 256;

  const __bf16* bp = hhT + (size_t)(nt * 16 + r) * DD + kbase + hk * 8;
  const int aoff   = (mt * 16 + r) * DD + kbase + hk * 8;
  const int col    = nt * 16 + r;
  const int mrow   = mt * 16 + hk * 8;

  unsigned* ctr = counters + mt * 32;     // one counter per mt-group, 128B apart

  // hh K-half strip resident in registers for all 512 steps (64 VGPRs)
  v16bf bfrag[8];
#pragma unroll
  for (int k = 0; k < 8; ++k) bfrag[k] = load_frag(bp + k * 32);

  // prefetch XW for t=0 (wave 0 only; it does the combine/store)
  float xwv[8] = {};
  if (widx == 0) {
    const __bf16* xw0 = XWb + (size_t)mrow * DD + col;
#pragma unroll
    for (int rr = 0; rr < 8; ++rr) xwv[rr] = (float)xw0[(size_t)rr * DD];
  }

  const __bf16* aprev = h0b + aoff;

  for (int t = 0; t < TT; ++t) {
    // ---- batched h_{t-1} fragment loads: issue ALL 16 b128 loads first ----
    v16bf afrag[8];
#pragma unroll
    for (int k = 0; k < 8; ++k) afrag[k] = load_frag(aprev + k * 32);

    // forbid the scheduler from sinking the loads into the WMMA chain
    __builtin_amdgcn_sched_barrier(0);

    // ---- 8-deep register WMMA chain over this wave's K-half ----
    v8f acc = {};
#pragma unroll
    for (int k = 0; k < 8; ++k) acc = wmma_bf16(afrag[k], bfrag[k], acc);

    // ---- cross-wave K-reduction through LDS ----
    if (widx == 1) part[lane] = acc;
    __syncthreads();

    const size_t base = (size_t)t * BB * DD;
    if (widx == 0) {
      v8f other = part[lane];
      __bf16* hp = H + base + (size_t)mrow * DD + col;
#pragma unroll
      for (int rr = 0; rr < 8; ++rr) {
        float v = acc[rr] + other[rr] + xwv[rr];
        hp[(size_t)rr * DD] = (__bf16)(v > 0.0f ? v : 0.0f);
      }
      // prefetch next step's XW while waiting at the barrier
      if (t + 1 < TT) {
        const __bf16* xwn = XWb + (size_t)(t + 1) * BB * DD + (size_t)mrow * DD + col;
#pragma unroll
        for (int rr = 0; rr < 8; ++rr) xwv[rr] = (float)xwn[(size_t)rr * DD];
      }
    }

    // ---- per-row-group generation barrier (32 blocks per group) ----
    __builtin_amdgcn_fence(__ATOMIC_RELEASE, "agent");   // make H[t] stores visible
    __syncthreads();
    if (threadIdx.x == 0) {
      __hip_atomic_fetch_add(ctr, 1u, __ATOMIC_RELAXED, __HIP_MEMORY_SCOPE_AGENT);
      const unsigned target = 32u * (unsigned)(t + 1);
      while (__hip_atomic_load(ctr, __ATOMIC_RELAXED, __HIP_MEMORY_SCOPE_AGENT) < target)
        __builtin_amdgcn_s_sleep(1);
    }
    __syncthreads();
    __builtin_amdgcn_fence(__ATOMIC_ACQUIRE, "agent");   // invalidate to see peers' H[t]

    aprev = H + base + aoff;
  }
}

// ---------- launcher ----------
extern "C" void kernel_launch(void* const* d_in, const int* in_sizes, int n_in,
                              void* d_out, int out_size, void* d_ws, size_t ws_size,
                              hipStream_t stream) {
  (void)in_sizes; (void)n_in; (void)out_size; (void)ws_size;
  const float* x  = (const float*)d_in[0];   // [T,B,512]
  const float* xh = (const float*)d_in[1];   // [512,512]
  const float* hh = (const float*)d_in[2];   // [512,512]
  const float* hb = (const float*)d_in[3];   // [512]
  const float* o  = (const float*)d_in[4];   // [512,512]
  const float* h0 = (const float*)d_in[5];   // [1,512]

  char* ws = (char*)d_ws;
  __bf16*   xhT      = (__bf16*)(ws + 0x0000000);  // 512 KB
  __bf16*   hhT      = (__bf16*)(ws + 0x0080000);  // 512 KB
  __bf16*   oT       = (__bf16*)(ws + 0x0100000);  // 512 KB
  __bf16*   h0b      = (__bf16*)(ws + 0x0180000);  // 64 KB
  unsigned* counters = (unsigned*)(ws + 0x0190000); // 4 counters, 128B apart
  __bf16*   xb       = (__bf16*)(ws + 0x01A0000);  // 32 MB  x in bf16
  __bf16*   XWb      = (__bf16*)(ws + 0x21A0000);  // 32 MB  x@xh + hb
  __bf16*   H        = (__bf16*)(ws + 0x41A0000);  // 32 MB  all h_t
  float*    Y        = (float*)d_out;              // [T,B,512] f32

  // prep: weight transposes + conversions
  cvt_transpose512<<<1024, 256, 0, stream>>>(xh, xhT);
  cvt_transpose512<<<1024, 256, 0, stream>>>(hh, hhT);
  cvt_transpose512<<<1024, 256, 0, stream>>>(o, oT);
  cvt_bf16<<<(TT * BB * DD) / 256, 256, 0, stream>>>(x, xb, TT * BB * DD);
  bcast_h0<<<(BB * DD) / 256, 256, 0, stream>>>(h0, h0b);
  hipMemsetAsync(counters, 0, 1024, stream);

  // XW = x @ xh + hb   (M = T*B = 32768)
  gemm512_bf16<true><<<2048, 256, 0, stream>>>(xb, xhT, hb, (void*)XWb);

  // sequential recurrence: 128 blocks x 2 waves (split-K) persistent
  rnn_steps<<<128, 64, 0, stream>>>(hhT, XWb, h0b, H, counters);

  // Y = H @ o
  gemm512_bf16<false><<<2048, 256, 0, stream>>>(H, oT, nullptr, (void*)Y);
}
